// PolicyValueNet_32323923870425
// MI455X (gfx1250) — matrix operand order repaired and checked by reading, not compiled, not run on hardware
//
#include <hip/hip_runtime.h>
#include <hip/hip_bf16.h>

typedef _Float16 half_t;
typedef __attribute__((ext_vector_type(16))) _Float16 v16h;
typedef __attribute__((ext_vector_type(8)))  _Float16 v8h;
typedef __attribute__((ext_vector_type(8)))  float    v8f;
typedef __attribute__((ext_vector_type(4))) unsigned int u32x4;
typedef __attribute__((ext_vector_type(8))) int          i32x8;
typedef __attribute__((ext_vector_type(4))) int          i32x4;

#define N_MOVES 200000
#define HW 400   // 20*20

// Tensor Data Mover availability (guarded: falls back to LDS staging loop)
#if defined(__has_builtin)
#if __has_builtin(__builtin_amdgcn_tensor_load_to_lds) && \
    __has_builtin(__builtin_amdgcn_s_wait_tensorcnt)
#define USE_TDM 1
#endif
#endif
#ifndef USE_TDM
#define USE_TDM 0
#endif

// ---------------------------------------------------------------------------
// Weight repack fp32 [CO][Cin][3][3] -> f16 [K/32][co][32], K zero-padded to 32
// ---------------------------------------------------------------------------
__global__ void wcvt_kernel(const float* __restrict__ w, half_t* __restrict__ wh, int Cin) {
    const int K = Cin * 9;
    const int Kpad = (K + 31) & ~31;
    int idx = blockIdx.x * 256 + threadIdx.x;
    if (idx >= Kpad * 128) return;
    int kblk = idx >> 12;          // / 4096
    int rem  = idx & 4095;
    int co   = rem >> 5;
    int kin  = rem & 31;
    int k    = kblk * 32 + kin;
    wh[idx] = (k < K) ? (half_t)w[co * K + k] : (half_t)0.0f;
}

// Policy W1 repack fp32 [128][147] -> f16 fragment layout [5][128][32] (K pad 160)
__global__ void wcvt_pol_kernel(const float* __restrict__ w1, half_t* __restrict__ w1h) {
    int e = blockIdx.x * 256 + threadIdx.x;
    if (e >= 5 * 4096) return;
    int kblk = e >> 12, rem = e & 4095;
    int j = rem >> 5, kin = rem & 31;
    int k = kblk * 32 + kin;
    w1h[e] = (k < 147) ? (half_t)w1[j * 147 + k] : (half_t)0.0f;
}

// ---------------------------------------------------------------------------
// 3x3 SAME conv as im2col GEMM with v_wmma_f32_16x16x32_f16.
// grid = 25 (M-tiles of 16 positions), block = 256 (8 waves = 8 N-tiles).
// x: [Cin][400] fp32, wh: repacked f16 weights, y: [128][400] fp32 (+bias)
// ---------------------------------------------------------------------------
__global__ void conv_wmma_kernel(const float* __restrict__ x, const half_t* __restrict__ wh,
                                 const float* __restrict__ bias, float* __restrict__ y,
                                 int Cin) {
    const int K    = Cin * 9;
    const int Kpad = (K + 31) & ~31;          // 96 (stem) or 1152
    const int AST  = 1160;                    // LDS row stride (halves); 2320B, 16B aligned
    __shared__ __align__(32) half_t Ah[16 * AST];

    const int tid   = threadIdx.x;
    const int mtile = blockIdx.x;             // 0..24

    // Stage im2col A tile: 16 positions x Kpad (zero-padded, SAME padding)
    for (int e = tid; e < 16 * Kpad; e += 256) {
        int row = e / Kpad, k = e - row * Kpad;
        half_t val = (half_t)0.0f;
        if (k < K) {
            int ci = k / 9, rs = k - ci * 9;
            int r = rs / 3, s = rs - r * 3;
            int p  = mtile * 16 + row;
            int py = p / 20 + r - 1;
            int px = p % 20 + s - 1;
            if (py >= 0 && py < 20 && px >= 0 && px < 20)
                val = (half_t)x[ci * HW + py * 20 + px];
        }
        Ah[row * AST + k] = val;
    }
    __syncthreads();

    const int wave = tid >> 5;                // N-tile 0..7
    const int lane = tid & 31;
    const int g    = lane >> 4;               // lane group
    const int mn   = lane & 15;               // A row / B col
    const int co   = wave * 16 + mn;

    v8f c = {};
    for (int k0 = 0; k0 < Kpad; k0 += 32) {
        // A fragment: contiguous k = 16*g + s, matching B
        v8h alo = *(const v8h*)&Ah[mn * AST + k0 + 16 * g];
        v8h ahi = *(const v8h*)&Ah[mn * AST + k0 + 16 * g + 8];
        v16h a;
#pragma unroll
        for (int h = 0; h < 8; ++h) { a[h] = alo[h]; a[8 + h] = ahi[h]; }
        // B fragment: one contiguous 32B chunk from repacked weights
        v16h b = *(const v16h*)&wh[(k0 >> 5) * 4096 + co * 32 + 16 * g];
        if (k0 + 32 < Kpad)
            __builtin_prefetch(&wh[((k0 >> 5) + 1) * 4096 + co * 32], 0, 1);
        c = __builtin_amdgcn_wmma_f32_16x16x32_f16(false, a, false, b, (short)0, c,
                                                   false, false);
    }

    // C layout: VGPR i -> row i + 8*g, col = mn
    const float bv = bias[co];
#pragma unroll
    for (int i = 0; i < 8; ++i) {
        int p = mtile * 16 + i + 8 * g;       // always < 400 (25*16 == 400)
        y[co * HW + p] = c[i] + bv;
    }
}

// ---------------------------------------------------------------------------
// GroupNorm over trunk (C=128, groups=8, 6400 elems/group) + optional
// residual-add + optional ReLU.  grid = 8, block = 256.
// ---------------------------------------------------------------------------
__global__ void gn_kernel(const float* __restrict__ yin, const float* __restrict__ scale,
                          const float* __restrict__ bias, const float* __restrict__ residual,
                          float* __restrict__ out, int relu) {
    const int grp = blockIdx.x;
    const float* base = yin + grp * 6400;
    const float* rb   = residual ? residual + grp * 6400 : nullptr;
    float* ob = out + grp * 6400;

    __shared__ float wred[8][2];
    __shared__ float mv[2];
    const int tid = threadIdx.x, wave = tid >> 5, lane = tid & 31;

    float s = 0.f, s2 = 0.f;
    for (int i = tid; i < 6400; i += 256) { float v = base[i]; s += v; s2 += v * v; }
    for (int m = 16; m >= 1; m >>= 1) {
        s  += __shfl_xor(s, m, 32);
        s2 += __shfl_xor(s2, m, 32);
    }
    if (lane == 0) { wred[wave][0] = s; wred[wave][1] = s2; }
    __syncthreads();
    if (tid == 0) {
        float ts = 0.f, t2 = 0.f;
        for (int w = 0; w < 8; ++w) { ts += wred[w][0]; t2 += wred[w][1]; }
        float mean = ts / 6400.f;
        float var  = t2 / 6400.f - mean * mean;
        mv[0] = mean;
        mv[1] = rsqrtf(var + 1e-5f);
    }
    __syncthreads();
    const float mean = mv[0], rstd = mv[1];
    for (int i = tid; i < 6400; i += 256) {
        int c = grp * 16 + i / HW;
        float v = (base[i] - mean) * rstd * scale[c] + bias[c];
        if (rb) v += rb[i];
        if (relu) v = v > 0.f ? v : 0.f;
        ob[i] = v;
    }
}

// ---------------------------------------------------------------------------
// Fused policy head: gather cell features -> feats(160, f16) -> WMMA GEMM with
// W1 (160x128 in LDS, DMA'd by the Tensor Data Mover) -> bias+ReLU -> dot with
// w2 -> logits.   grid = 3125 blocks x 64 moves, block = 128 (4 waves).
// ---------------------------------------------------------------------------
__global__ void policy_kernel(const float* __restrict__ fmap, const int* __restrict__ cells,
                              const int* __restrict__ cell_counts,
                              const int* __restrict__ piece_ids, const float* __restrict__ emb,
                              const float* __restrict__ anchors, const float* __restrict__ szv,
                              const half_t* __restrict__ w1h, const float* __restrict__ pol_w1,
                              const float* __restrict__ pol_b1,
                              const float* __restrict__ pol_w2, const float* __restrict__ pol_b2,
                              float* __restrict__ logits) {
    const int FST = 168;                               // feats LDS row stride (halves)
    __shared__ __align__(32) half_t W1s[5 * 128 * 32]; // [kblk][j][32] = 40KB
    __shared__ __align__(32) half_t featsh[64 * FST];  // 21KB
    __shared__ float b1s[128];
    __shared__ float w2s[128];

    const int tid = threadIdx.x;                       // 128 threads
    const int m0  = blockIdx.x * 64;

#if USE_TDM
    // Issue a 40960B flat TDM copy (global f16 W1, already repacked) into LDS.
    // One issue per block (wave 0); overlaps with the gather below.
    if (tid == 0) {
        unsigned long long ga = (unsigned long long)(uintptr_t)w1h;
        unsigned lds_off = (unsigned)(unsigned long long)(uintptr_t)&W1s[0];
        u32x4 g0 = {};
        g0[0] = 1u;                                         // count=1, user D#
        g0[1] = lds_off;                                    // lds_addr
        g0[2] = (unsigned)(ga & 0xffffffffu);               // global_addr[31:0]
        g0[3] = (unsigned)((ga >> 32) & 0x1ffffffu) | (2u << 30); // addr[56:32]|type=2
        i32x8 g1 = {};
        g1[0] = (3 << 16);                                  // data_size = 8 bytes
        g1[1] = (int)((5120u & 0xffffu) << 16);             // tensor_dim0 lo (5120 elems)
        g1[2] = (int)((1u & 0xffffu) << 16);                // dim0 hi=0 | tensor_dim1=1
        g1[3] = (int)(5120u << 16);                         // dim1 hi=0 | tile_dim0=5120
        g1[4] = 0;                                          // tile_dim1/2 unused
        g1[5] = 5120;                                       // tensor_dim0_stride
        g1[6] = 0;
        g1[7] = 0;
        i32x4 z4 = {};
#if defined(__clang_major__) && __clang_major__ >= 23
        i32x8 z8 = {};
        __builtin_amdgcn_tensor_load_to_lds(g0, g1, z4, z4, z8, 0);
#else
        __builtin_amdgcn_tensor_load_to_lds(g0, g1, z4, z4, 0);
#endif
    }
#else
    // Fallback: stage W1 (already repacked f16) with a copy loop
    for (int e = tid; e < 5 * 4096; e += 128) W1s[e] = w1h[e];
#endif
    if (tid < 128) { b1s[tid] = pol_b1[tid]; w2s[tid] = pol_w2[tid]; }

    // Gather + build feats tile (64 moves x 160 dims, f16) while the DMA runs
    for (int e = tid; e < 64 * 160; e += 128) {
        int mi = e / 160, f = e - mi * 160;
        int m = m0 + mi;
        float v = 0.f;
        if (m < N_MOVES) {
            if (f < 128) {
                int cnt = cell_counts[m];
                float acc = 0.f;
#pragma unroll
                for (int k = 0; k < 5; ++k) {
                    if (k < cnt) {
                        int cx = cells[(m * 5 + k) * 2 + 0];
                        int cy = cells[(m * 5 + k) * 2 + 1];
                        acc += fmap[f * HW + cy * 20 + cx];
                    }
                }
                v = acc / (float)cnt;
            } else if (f < 144) {
                v = emb[piece_ids[m] * 16 + (f - 128)];
            } else if (f < 146) {
                v = anchors[m * 2 + (f - 144)];
            } else if (f == 146) {
                v = szv[m];
            }
        }
        featsh[mi * FST + f] = (half_t)v;
    }
#if USE_TDM
    if (tid == 0) __builtin_amdgcn_s_wait_tensorcnt(0);   // DMA complete (wave 0)
#endif
    __syncthreads();

    const int wave = tid >> 5;                 // M-tile 0..3
    const int lane = tid & 31;
    const int g    = lane >> 4;
    const int mn   = lane & 15;
    const int mi0  = wave * 16;

    float lp[8] = {0.f, 0.f, 0.f, 0.f, 0.f, 0.f, 0.f, 0.f};
    for (int nt = 0; nt < 8; ++nt) {
        v8f c = {};
#pragma unroll
        for (int k0 = 0; k0 < 160; k0 += 32) {
            v8h alo = *(const v8h*)&featsh[(mi0 + mn) * FST + k0 + 16 * g];
            v8h ahi = *(const v8h*)&featsh[(mi0 + mn) * FST + k0 + 16 * g + 8];
            v16h a;
#pragma unroll
            for (int h = 0; h < 8; ++h) { a[h] = alo[h]; a[8 + h] = ahi[h]; }
            v16h b = *(const v16h*)&W1s[(k0 >> 5) * 4096 + (nt * 16 + mn) * 32 + 16 * g];
            c = __builtin_amdgcn_wmma_f32_16x16x32_f16(false, a, false, b, (short)0, c,
                                                       false, false);
        }
        const int n = nt * 16 + mn;
        const float bn = b1s[n], wn = w2s[n];
#pragma unroll
        for (int i = 0; i < 8; ++i) {
            float h = c[i] + bn;
            h = h > 0.f ? h : 0.f;
            lp[i] += h * wn;                   // fused hmlp @ w2
        }
    }
    // Reduce over the 16 columns held by each 16-lane half
#pragma unroll
    for (int i = 0; i < 8; ++i) {
        lp[i] += __shfl_xor(lp[i], 1, 32);
        lp[i] += __shfl_xor(lp[i], 2, 32);
        lp[i] += __shfl_xor(lp[i], 4, 32);
        lp[i] += __shfl_xor(lp[i], 8, 32);
    }
    if (mn == 0) {
        const float pb2 = pol_b2[0];
#pragma unroll
        for (int i = 0; i < 8; ++i) {
            int m = m0 + mi0 + i + 8 * g;
            if (m < N_MOVES) logits[m] = lp[i] + pb2;
        }
    }
}

// ---------------------------------------------------------------------------
// Value head: 1x1 conv 128->64, GN(8 groups), ReLU, mean-pool, MLP 107->256->
// 128->1, tanh.  Single block of 256 threads (negligible FLOPs).
// ---------------------------------------------------------------------------
__global__ void value_kernel(const float* __restrict__ fmap, const float* __restrict__ cw,
                             const float* __restrict__ cb, const float* __restrict__ gs,
                             const float* __restrict__ gb, const float* __restrict__ srem,
                             const float* __restrict__ orem, const float* __restrict__ phase,
                             const float* __restrict__ w1, const float* __restrict__ b1,
                             const float* __restrict__ w2, const float* __restrict__ b2,
                             const float* __restrict__ w3, const float* __restrict__ b3,
                             float* __restrict__ outv) {
    __shared__ half_t vsh[64 * HW];            // 51.2KB
    __shared__ float stats[8][2];
    __shared__ float pooled[64];
    __shared__ float vin[112];
    __shared__ float h1[256];
    __shared__ float h2[128];
    const int tid = threadIdx.x, wave = tid >> 5, lane = tid & 31;

    // 1x1 conv
    for (int e = tid; e < 64 * HW; e += 256) {
        int c = e / HW, p = e - c * HW;
        float acc = cb[c];
        for (int ci = 0; ci < 128; ++ci) acc += cw[c * 128 + ci] * fmap[ci * HW + p];
        vsh[e] = (half_t)acc;
    }
    __syncthreads();

    // GN stats: wave g reduces group g (8 channels x 400)
    {
        float s = 0.f, s2 = 0.f;
        for (int i = lane; i < 3200; i += 32) {
            float v = (float)vsh[wave * 3200 + i];
            s += v; s2 += v * v;
        }
        for (int m = 16; m >= 1; m >>= 1) {
            s  += __shfl_xor(s, m, 32);
            s2 += __shfl_xor(s2, m, 32);
        }
        if (lane == 0) {
            float mean = s / 3200.f;
            float var  = s2 / 3200.f - mean * mean;
            stats[wave][0] = mean;
            stats[wave][1] = rsqrtf(var + 1e-5f);
        }
    }
    __syncthreads();

    // normalize + ReLU + mean-pool
    if (tid < 64) {
        int c = tid, grp = c >> 3;
        float mean = stats[grp][0], rstd = stats[grp][1];
        float sc = gs[c], bi = gb[c], acc = 0.f;
        for (int p = 0; p < HW; ++p) {
            float v = ((float)vsh[c * HW + p] - mean) * rstd * sc + bi;
            acc += v > 0.f ? v : 0.f;
        }
        pooled[c] = acc * (1.f / 400.f);
    }
    __syncthreads();
    if (tid < 64)        vin[tid] = pooled[tid];
    else if (tid < 85)   vin[tid] = srem[tid - 64];
    else if (tid < 106)  vin[tid] = orem[tid - 85];
    else if (tid == 106) vin[tid] = phase[0];
    __syncthreads();
    {
        float acc = b1[tid];
        for (int k = 0; k < 107; ++k) acc += w1[tid * 107 + k] * vin[k];
        h1[tid] = acc > 0.f ? acc : 0.f;
    }
    __syncthreads();
    if (tid < 128) {
        float acc = b2[tid];
        for (int k = 0; k < 256; ++k) acc += w2[tid * 256 + k] * h1[k];
        h2[tid] = acc > 0.f ? acc : 0.f;
    }
    __syncthreads();
    if (tid == 0) {
        float acc = b3[0];
        for (int k = 0; k < 128; ++k) acc += w3[k] * h2[k];
        outv[0] = tanhf(acc);
    }
}

// ---------------------------------------------------------------------------
extern "C" void kernel_launch(void* const* d_in, const int* in_sizes, int n_in,
                              void* d_out, int out_size, void* d_ws, size_t ws_size,
                              hipStream_t stream) {
    const float* board   = (const float*)d_in[0];
    const float* srem    = (const float*)d_in[1];
    const float* orem    = (const float*)d_in[2];
    const float* phase   = (const float*)d_in[3];
    const float* anchors = (const float*)d_in[4];
    const float* szv     = (const float*)d_in[5];
    const float* stem_w  = (const float*)d_in[6];
    const float* stem_b  = (const float*)d_in[7];
    const float* stem_gs = (const float*)d_in[8];
    const float* stem_gb = (const float*)d_in[9];
    const float* blk_w   = (const float*)d_in[10];
    const float* blk_b   = (const float*)d_in[11];
    const float* blk_gs  = (const float*)d_in[12];
    const float* blk_gb  = (const float*)d_in[13];
    const float* emb     = (const float*)d_in[14];
    const float* pol_w1  = (const float*)d_in[15];
    const float* pol_b1  = (const float*)d_in[16];
    const float* pol_w2  = (const float*)d_in[17];
    const float* pol_b2  = (const float*)d_in[18];
    const float* val_cw  = (const float*)d_in[19];
    const float* val_cb  = (const float*)d_in[20];
    const float* val_gs  = (const float*)d_in[21];
    const float* val_gb  = (const float*)d_in[22];
    const float* val_w1  = (const float*)d_in[23];
    const float* val_b1  = (const float*)d_in[24];
    const float* val_w2  = (const float*)d_in[25];
    const float* val_b2  = (const float*)d_in[26];
    const float* val_w3  = (const float*)d_in[27];
    const float* val_b3  = (const float*)d_in[28];
    const int* piece_ids   = (const int*)d_in[29];
    const int* cells       = (const int*)d_in[30];
    const int* cell_counts = (const int*)d_in[31];

    // Workspace layout (fp32 feature maps + f16 repacked weights)
    float*  xA  = (float*)d_ws;                  // 128*400
    float*  xB  = xA + 128 * HW;
    float*  xC  = xB + 128 * HW;
    half_t* wh  = (half_t*)(xC + 128 * HW);      // 1152*128 halves (conv weights)
    half_t* w1h = wh + 1152 * 128;               // 5*4096 halves (policy W1, TDM source)

    float* logits = (float*)d_out;
    float* outv   = logits + N_MOVES;

    // ---- stem: conv(8->128) + GN + ReLU ----
    wcvt_kernel<<<(96 * 128 + 255) / 256, 256, 0, stream>>>(stem_w, wh, 8);
    conv_wmma_kernel<<<25, 256, 0, stream>>>(board, wh, stem_b, xB, 8);
    gn_kernel<<<8, 256, 0, stream>>>(xB, stem_gs, stem_gb, nullptr, xA, 1);

    // ---- 10 residual blocks ----
    for (int i = 0; i < 10; ++i) {
        const float* w0  = blk_w  + (size_t)(i * 2 + 0) * 128 * 128 * 9;
        const float* w1p = blk_w  + (size_t)(i * 2 + 1) * 128 * 128 * 9;
        const float* b0  = blk_b  + (i * 2 + 0) * 128;
        const float* b1p = blk_b  + (i * 2 + 1) * 128;
        const float* gs0 = blk_gs + (i * 2 + 0) * 128;
        const float* gs1 = blk_gs + (i * 2 + 1) * 128;
        const float* gb0 = blk_gb + (i * 2 + 0) * 128;
        const float* gb1 = blk_gb + (i * 2 + 1) * 128;

        wcvt_kernel<<<(1152 * 128 + 255) / 256, 256, 0, stream>>>(w0, wh, 128);
        conv_wmma_kernel<<<25, 256, 0, stream>>>(xA, wh, b0, xB, 128);
        gn_kernel<<<8, 256, 0, stream>>>(xB, gs0, gb0, nullptr, xB, 1);

        wcvt_kernel<<<(1152 * 128 + 255) / 256, 256, 0, stream>>>(w1p, wh, 128);
        conv_wmma_kernel<<<25, 256, 0, stream>>>(xB, wh, b1p, xC, 128);
        gn_kernel<<<8, 256, 0, stream>>>(xC, gs1, gb1, xA, xA, 1);  // +res, ReLU, in-place
    }

    // ---- policy head (dominant GEMM, WMMA + TDM-staged W1) ----
    wcvt_pol_kernel<<<(5 * 4096 + 255) / 256, 256, 0, stream>>>(pol_w1, w1h);
    policy_kernel<<<(N_MOVES + 63) / 64, 128, 0, stream>>>(
        xA, cells, cell_counts, piece_ids, emb, anchors, szv,
        w1h, pol_w1, pol_b1, pol_w2, pol_b2, logits);

    // ---- value head ----
    value_kernel<<<1, 256, 0, stream>>>(xA, val_cw, val_cb, val_gs, val_gb,
                                        srem, orem, phase,
                                        val_w1, val_b1, val_w2, val_b2,
                                        val_w3, val_b3, outv);
}
